// CausalMultiHeadSelfAttention_798863917078
// MI455X (gfx1250) — matrix operand order repaired and checked by reading, not compiled, not run on hardware
//
#include <hip/hip_runtime.h>

typedef _Float16 half_t;
typedef __attribute__((ext_vector_type(16))) _Float16 v16h;
typedef __attribute__((ext_vector_type(8)))  _Float16 v8h;
typedef __attribute__((ext_vector_type(8)))  float    v8f;
typedef __attribute__((ext_vector_type(4)))  float    v4f;
typedef __attribute__((ext_vector_type(4)))  int      v4i;

#define DMODEL 1024
#define NHEADS 16
#define DK     64
#define BB     4
#define TT     2048
#define MTOT   (BB * TT)

static __device__ __forceinline__ v8f zero8() {
  v8f r;
#pragma unroll
  for (int i = 0; i < 8; ++i) r[i] = 0.f;
  return r;
}

static __device__ __forceinline__ v16h cat8(v8h lo, v8h hi) {
  return __builtin_shufflevector(lo, hi, 0, 1, 2, 3, 4, 5, 6, 7,
                                 8, 9, 10, 11, 12, 13, 14, 15);
}

// ---- CDNA5 async global->LDS staging (ASYNCcnt path), with portable fallback.
static __device__ __forceinline__ void async_cp16(const half_t* g, half_t* l) {
#if __has_builtin(__builtin_amdgcn_global_load_async_to_lds_b128)
  auto gp = (__attribute__((address_space(1))) v4i*)(uintptr_t)(const void*)g;
  auto lp = (__attribute__((address_space(3))) v4i*)(unsigned int)(uintptr_t)(void*)l;
  __builtin_amdgcn_global_load_async_to_lds_b128(gp, lp, 0, 0);
#else
  *(v8h*)l = *(const v8h*)g;
#endif
}

static __device__ __forceinline__ void wait_async0() {
#if __has_builtin(__builtin_amdgcn_s_wait_asynccnt)
  __builtin_amdgcn_s_wait_asynccnt(0);
#else
  asm volatile("s_wait_asynccnt 0" ::: "memory");
#endif
}

// -----------------------------------------------------------------------------
// One-time f32 -> f16 repack (8 elements/thread, b128 in / b128 out).
// -----------------------------------------------------------------------------
__global__ __launch_bounds__(256)
void cvt_f32_f16(const float* __restrict__ src, half_t* __restrict__ dst, int n8)
{
  const int i = blockIdx.x * blockDim.x + threadIdx.x;
  if (i >= n8) return;
  const size_t off = (size_t)i * 8;
  const v4f a = *(const v4f*)(src + off);
  const v4f b = *(const v4f*)(src + off + 4);
  v8h r;
#pragma unroll
  for (int c = 0; c < 4; ++c) { r[c] = (half_t)a[c]; r[4 + c] = (half_t)b[c]; }
  *(v8h*)(dst + off) = r;
}

// -----------------------------------------------------------------------------
// Y[m,n] = sum_k A[m,k] * W[n,k], all-f16 operands, f32 WMMA accumulate.
// Block = 8 waves -> 128x64 tile; 64x32 W tile double-buffered in LDS via
// async-to-LDS; A fragments straight from global (per-wave rows, K contiguous).
// -----------------------------------------------------------------------------
template <typename OutT, bool HEADMAJ>
__global__ __launch_bounds__(256)
void gemm_h(const half_t* __restrict__ A, const half_t* __restrict__ W,
            OutT* __restrict__ Y, int M, int N, int K)
{
  __shared__ __align__(16) half_t wtile[2][64][32];
  const int tid  = threadIdx.x;
  const int lane = tid & 31;
  const int wv   = tid >> 5;
  const int lr   = lane & 15;
  const int hs   = lane >> 4;
  const int m0   = blockIdx.x * 128 + wv * 16;
  const int n0   = blockIdx.y * 64;

  const int sr = tid >> 2;
  const int sc = (tid & 3) * 8;
  const half_t* wsrc = W + (size_t)(n0 + sr) * K + sc;

  v8f acc[4];
#pragma unroll
  for (int j = 0; j < 4; ++j) acc[j] = zero8();

  const half_t* arow = A + (size_t)(m0 + lr) * K;

  async_cp16(wsrc, &wtile[0][sr][sc]);
  for (int k0 = 0; k0 < K; k0 += 32) {
    const int buf = (k0 >> 5) & 1;
    wait_async0();
    __syncthreads();
    if (k0 + 32 < K)
      async_cp16(wsrc + k0 + 32, &wtile[buf ^ 1][sr][sc]);

    const half_t* pa = arow + k0 + hs * 16;
    const v16h a = cat8(*(const v8h*)pa, *(const v8h*)(pa + 8));
    __builtin_prefetch(arow + k0 + 64, 0, 1);
#pragma unroll
    for (int j = 0; j < 4; ++j) {
      const half_t* pb = &wtile[buf][j * 16 + lr][hs * 16];
      const v16h bm = cat8(*(const v8h*)pb, *(const v8h*)(pb + 8));
      acc[j] = __builtin_amdgcn_wmma_f32_16x16x32_f16(
          false, a, false, bm, (short)0, acc[j], false, false);
    }
  }

#pragma unroll
  for (int j = 0; j < 4; ++j)
#pragma unroll
    for (int v = 0; v < 8; ++v) {
      const int m = m0 + v + 8 * hs;
      const int n = n0 + j * 16 + lr;
      if constexpr (HEADMAJ) {
        const int b = m >> 11, t = m & (TT - 1);
        const int h = n >> 6, d = n & (DK - 1);
        Y[(((size_t)(b * NHEADS + h) * TT) + t) * DK + d] = (OutT)acc[j][v];
      } else {
        Y[(size_t)m * N + n] = (OutT)acc[j][v];
      }
    }
}

// -----------------------------------------------------------------------------
// In-place RoPE on head-major f16 Q and K.
// -----------------------------------------------------------------------------
__global__ __launch_bounds__(256)
void rope_inplace(half_t* __restrict__ Qh, half_t* __restrict__ Kh,
                  const long long* __restrict__ pos)
{
  const int idx = blockIdx.x * blockDim.x + threadIdx.x;
  const int i  = idx & 31;
  const int t  = (idx >> 5) & (TT - 1);
  const int bh = idx >> 16;
  if (bh >= BB * NHEADS) return;

  const size_t base = ((size_t)bh * TT + t) * DK + 2 * i;
  const int b = bh >> 4;

  const float p   = (float)pos[b * TT + t];
  const float ang = p * __powf(10000.f, -(float)i / 32.f);
  float s, c;
  __sincosf(ang, &s, &c);

  const float q1 = (float)Qh[base], q2 = (float)Qh[base + 1];
  const float k1 = (float)Kh[base], k2 = (float)Kh[base + 1];
  Qh[base]     = (half_t)(q1 * c - q2 * s);
  Qh[base + 1] = (half_t)(q1 * s + q2 * c);
  Kh[base]     = (half_t)(k1 * c - k2 * s);
  Kh[base + 1] = (half_t)(k1 * s + k2 * c);
}

// -----------------------------------------------------------------------------
// Flash attention, block-cooperative K/V staging:
// Block = 8 waves = 128 consecutive queries of one (b,h). The block iterates a
// uniform key range; each 32-key K tile and V tile is async-staged into LDS
// (double-buffered) and shared by all 8 waves. Waves past their causal limit
// skip compute via a wave-uniform branch (EXEC stays all-1s for WMMA).
// -----------------------------------------------------------------------------
__global__ __launch_bounds__(256)
void flash_attn(const half_t* __restrict__ Qh, const half_t* __restrict__ Kh,
                const half_t* __restrict__ Vh, half_t* __restrict__ Hh)
{
  __shared__ __align__(16) half_t ktile[2][32][DK];   // 8 KB
  __shared__ __align__(16) half_t vtile[2][32][DK];   // 8 KB
  __shared__ __align__(16) half_t pbuf[8][16 * 32];   // 8 KB
  const int tid  = threadIdx.x;
  const int lane = tid & 31;
  const int wv   = tid >> 5;
  const int lr   = lane & 15;
  const int hs   = lane >> 4;
  const int bh   = blockIdx.x;
  const int qbase = blockIdx.y * 128;
  const int qt   = (qbase >> 4) + wv;
  const int b    = bh >> 4;
  const int h    = bh & 15;

  const half_t* Qp = Qh + (size_t)bh * TT * DK;
  const half_t* Kp = Kh + (size_t)bh * TT * DK;
  const half_t* Vp = Vh + (size_t)bh * TT * DK;

  // cooperative stage coords: 256 threads x 16B = one 32x64-half tile
  const int srow = tid >> 3;          // key row 0..31
  const int schk = (tid & 7) * 8;     // half offset in row

  // per-wave Q fragments
  v16h qa0, qa1;
  {
    const half_t* qrow = Qp + (size_t)(qt * 16 + lr) * DK;
    const half_t* p0 = qrow + hs * 16;
    qa0 = cat8(*(const v8h*)p0, *(const v8h*)(p0 + 8));
    const half_t* p1 = qrow + 32 + hs * 16;
    qa1 = cat8(*(const v8h*)p1, *(const v8h*)(p1 + 8));
  }

  v8f o[4];
#pragma unroll
  for (int j = 0; j < 4; ++j) o[j] = zero8();
  float mrow[8], lrow[8];
#pragma unroll
  for (int v = 0; v < 8; ++v) { mrow[v] = -1e30f; lrow[v] = 0.f; }

  const int qlast = qt * 16 + 15;           // wave-uniform causal limit
  const int nkb   = (qbase + 128) >> 5;     // uniform #key-slabs for the block

  // prologue stage slab 0
  async_cp16(Kp + (size_t)srow * DK + schk, &ktile[0][srow][schk]);
  async_cp16(Vp + (size_t)srow * DK + schk, &vtile[0][srow][schk]);

  for (int kb = 0; kb < nkb; ++kb) {
    const int kbase = kb << 5;
    const int buf = kb & 1;
    wait_async0();
    __syncthreads();                        // staged slab visible; prev reads done
    if (kb + 1 < nkb) {
      const size_t nb = (size_t)(kbase + 32 + srow) * DK + schk;
      async_cp16(Kp + nb, &ktile[buf ^ 1][srow][schk]);
      async_cp16(Vp + nb, &vtile[buf ^ 1][srow][schk]);
    }

    if (kbase <= qlast) {                   // wave-uniform: EXEC all-1s inside
      // ---- S = Q K^T from LDS K tile
      v8f s0 = zero8(), s1 = zero8();
      {
        const half_t* r0 = &ktile[buf][lr][hs * 16];
        const half_t* r1 = &ktile[buf][16 + lr][hs * 16];
        v16h bf = cat8(*(const v8h*)r0, *(const v8h*)(r0 + 8));
        s0 = __builtin_amdgcn_wmma_f32_16x16x32_f16(false, qa0, false, bf, (short)0, s0, false, false);
        bf = cat8(*(const v8h*)(r0 + 32), *(const v8h*)(r0 + 40));
        s0 = __builtin_amdgcn_wmma_f32_16x16x32_f16(false, qa1, false, bf, (short)0, s0, false, false);
        bf = cat8(*(const v8h*)r1, *(const v8h*)(r1 + 8));
        s1 = __builtin_amdgcn_wmma_f32_16x16x32_f16(false, qa0, false, bf, (short)0, s1, false, false);
        bf = cat8(*(const v8h*)(r1 + 32), *(const v8h*)(r1 + 40));
        s1 = __builtin_amdgcn_wmma_f32_16x16x32_f16(false, qa1, false, bf, (short)0, s1, false, false);
      }

      // ---- online softmax across the 32-key slab
      const float scale = 0.125f;   // 1/sqrt(64)
      half_t* pl = pbuf[wv];
#pragma unroll
      for (int v = 0; v < 8; ++v) {
        const int row = qt * 16 + v + 8 * hs;
        float a0 = s0[v] * scale;
        float a1 = s1[v] * scale;
        if (kbase + lr      > row) a0 = -1e30f;
        if (kbase + 16 + lr > row) a1 = -1e30f;
        float mv = fmaxf(a0, a1);
#pragma unroll
        for (int off = 1; off < 16; off <<= 1)
          mv = fmaxf(mv, __shfl_xor(mv, off, 32));
        const float mnew  = fmaxf(mrow[v], mv);
        const float alpha = __expf(mrow[v] - mnew);
        const float p0 = __expf(a0 - mnew);
        const float p1 = __expf(a1 - mnew);
        float rs = p0 + p1;
#pragma unroll
        for (int off = 1; off < 16; off <<= 1)
          rs += __shfl_xor(rs, off, 32);
        lrow[v] = lrow[v] * alpha + rs;
        mrow[v] = mnew;
#pragma unroll
        for (int j = 0; j < 4; ++j) o[j][v] *= alpha;
        pl[(v + 8 * hs) * 32 + lr]      = (half_t)p0;
        pl[(v + 8 * hs) * 32 + 16 + lr] = (half_t)p1;
      }
      asm volatile("s_wait_dscnt 0" ::: "memory");

      // ---- reload P as an A fragment (16 x 32 keys)
      v16h pa;
      {
        const half_t* pr = pl + lr * 32 + hs * 16;
        pa = cat8(*(const v8h*)pr, *(const v8h*)(pr + 8));
      }

      // ---- O += P V from LDS V tile (B frag: comps = keys, strided in LDS)
#pragma unroll
      for (int j = 0; j < 4; ++j) {
        const half_t* vb = &vtile[buf][hs * 16][j * 16 + lr];
        v16h vf;
#pragma unroll
        for (int c = 0; c < 16; ++c) vf[c] = vb[c * DK];
        o[j] = __builtin_amdgcn_wmma_f32_16x16x32_f16(
            false, pa, false, vf, (short)0, o[j], false, false);
      }
    }
  }

#pragma unroll
  for (int v = 0; v < 8; ++v) {
    const float invl = 1.f / lrow[v];
    const int t = qt * 16 + v + 8 * hs;
    half_t* dst = Hh + ((size_t)b * TT + t) * DMODEL + (size_t)h * DK;
#pragma unroll
    for (int j = 0; j < 4; ++j) dst[j * 16 + lr] = (half_t)(o[j][v] * invl);
  }
}

// -----------------------------------------------------------------------------
extern "C" void kernel_launch(void* const* d_in, const int* in_sizes, int n_in,
                              void* d_out, int out_size, void* d_ws, size_t ws_size,
                              hipStream_t stream)
{
  const float*     x   = (const float*)d_in[0];
  const long long* pos = (const long long*)d_in[1];
  const float*     Wq  = (const float*)d_in[2];
  const float*     Wk  = (const float*)d_in[3];
  const float*     Wv  = (const float*)d_in[4];
  const float*     Wo  = (const float*)d_in[5];
  float*           out = (float*)d_out;

  half_t* Xh  = (half_t*)d_ws;
  half_t* Whq = Xh  + (size_t)MTOT * DMODEL;
  half_t* Whk = Whq + (size_t)DMODEL * DMODEL;
  half_t* Whv = Whk + (size_t)DMODEL * DMODEL;
  half_t* Who = Whv + (size_t)DMODEL * DMODEL;
  half_t* Qh  = Who + (size_t)DMODEL * DMODEL;
  half_t* Kh  = Qh  + (size_t)MTOT * DMODEL;
  half_t* Vh  = Kh  + (size_t)MTOT * DMODEL;
  half_t* Hh  = Vh  + (size_t)MTOT * DMODEL;

  cvt_f32_f16<<<(MTOT * DMODEL / 8) / 256, 256, 0, stream>>>(x, Xh, MTOT * DMODEL / 8);
  const int w8 = DMODEL * DMODEL / 8;
  cvt_f32_f16<<<w8 / 256, 256, 0, stream>>>(Wq, Whq, w8);
  cvt_f32_f16<<<w8 / 256, 256, 0, stream>>>(Wk, Whk, w8);
  cvt_f32_f16<<<w8 / 256, 256, 0, stream>>>(Wv, Whv, w8);
  cvt_f32_f16<<<w8 / 256, 256, 0, stream>>>(Wo, Who, w8);

  const dim3 gg(MTOT / 128, DMODEL / 64);
  gemm_h<half_t, true><<<gg, 256, 0, stream>>>(Xh, Whq, Qh, MTOT, DMODEL, DMODEL);
  gemm_h<half_t, true><<<gg, 256, 0, stream>>>(Xh, Whk, Kh, MTOT, DMODEL, DMODEL);
  gemm_h<half_t, true><<<gg, 256, 0, stream>>>(Xh, Whv, Vh, MTOT, DMODEL, DMODEL);

  const int nrot = BB * NHEADS * TT * (DK / 2);
  rope_inplace<<<nrot / 256, 256, 0, stream>>>(Qh, Kh, pos);

  const dim3 ga(BB * NHEADS, TT / 128);
  flash_attn<<<ga, 256, 0, stream>>>(Qh, Kh, Vh, Hh);

  gemm_h<float, false><<<gg, 256, 0, stream>>>(Hh, Who, out, MTOT, DMODEL, DMODEL);
}
